// AttentionLayer_70617852281155
// MI455X (gfx1250) — matrix operand order repaired and checked by reading, not compiled, not run on hardware
//
#include <hip/hip_runtime.h>

#define D_DIM 1024
#define S_DIM 2048
#define B_DIM 8

typedef __bf16 bf16_t;
typedef __attribute__((ext_vector_type(16))) __bf16 v16bf;
typedef __attribute__((ext_vector_type(8)))  __bf16 v8bf;
typedef __attribute__((ext_vector_type(4)))  __bf16 v4bf;
typedef __attribute__((ext_vector_type(8)))  float   v8f;
typedef __attribute__((ext_vector_type(4)))  float   v4f;

__device__ __forceinline__ bf16_t f2bf(float f) {
  union { float f; unsigned u; } a; a.f = f;
  unsigned r = a.u + 0x7FFFu + ((a.u >> 16) & 1u);   // round-to-nearest-even
  unsigned short hs = (unsigned short)(r >> 16);
  return __builtin_bit_cast(bf16_t, hs);
}

__device__ __forceinline__ v8f zero8() { v8f z = {0.f,0.f,0.f,0.f,0.f,0.f,0.f,0.f}; return z; }

// A-matrix fragment (16x32 bf16): lane holds row m=lane&15,
// K chunks [8h, 8h+8) and [16+8h, 16+8h+8), h = lane>>4.
__device__ __forceinline__ v16bf load_a_frag(const bf16_t* base, int ld, int k0, int lane) {
  const int r = lane & 15;
  const int h = (lane >> 4) << 3;           // 0 or 8
  const bf16_t* p = base + (size_t)r * ld + k0 + h;
  v8bf lo = *(const v8bf*)(p);
  v8bf hi = *(const v8bf*)(p + 16);
  return __builtin_shufflevector(lo, hi, 0,1,2,3,4,5,6,7,8,9,10,11,12,13,14,15);
}

// B-matrix fragment (32x16 bf16): lane holds column n=col0+(lane&15),
// K chunks [k0+8h, k0+8h+8) and [k0+16+8h, k0+16+8h+8), h = lane>>4.
// `base` is B^T stored row-major (row = column of B), row stride ld.
__device__ __forceinline__ v16bf load_b_frag(const bf16_t* base, int ld, int col0, int k0, int lane) {
  const int n = col0 + (lane & 15);
  const int h = (lane >> 4) << 3;           // 0 or 8
  const bf16_t* p = base + (size_t)n * ld + k0 + h;
  v8bf lo = *(const v8bf*)(p);
  v8bf hi = *(const v8bf*)(p + 16);
  return __builtin_shufflevector(lo, hi, 0,1,2,3,4,5,6,7,8,9,10,11,12,13,14,15);
}

__device__ __forceinline__ v8f wmma_bf16(v16bf a, v16bf b, v8f c) {
  return __builtin_amdgcn_wmma_f32_16x16x32_bf16(false, a, false, b, (short)0, c, false, false);
}

// ---------------------------------------------------------------- kernel 1
__global__ void attn_convert_w(const float* __restrict__ wq, const float* __restrict__ wk,
                               const float* __restrict__ wv, bf16_t* __restrict__ wb) {
  int i = blockIdx.x * blockDim.x + threadIdx.x;
  const int n = D_DIM * D_DIM;
  if (i < n) {
    wb[i]         = f2bf(wq[i]);
    wb[i + n]     = f2bf(wk[i]);
    wb[i + 2 * n] = f2bf(wv[i]);
  }
}

// ---------------------------------------------------------------- kernel 2
// grid: (BS/16, 3). Each block: one 16-token tile of one projection.
// Wave w handles output columns [w*128, w*128+128).
__global__ __launch_bounds__(256) void attn_qkv(
    const float* __restrict__ x, const bf16_t* __restrict__ wb,
    const float* __restrict__ bq, const float* __restrict__ bk, const float* __restrict__ bv,
    bf16_t* __restrict__ qb, bf16_t* __restrict__ kb, bf16_t* __restrict__ vT)
{
  const int mt   = blockIdx.x;
  const int mat  = blockIdx.y;
  const int tid  = threadIdx.x;
  const int lane = tid & 31;
  const int wid  = tid >> 5;
  const int half = lane >> 4;
  const int nl   = lane & 15;
  const int tok0 = mt * 16;

  __shared__ bf16_t sX[16 * 1032];           // x tile, bf16, padded stride

  // stage + convert x tile: 16 x 1024 fp32 -> bf16 LDS
  for (int it = 0; it < 16; ++it) {
    int idx = tid + it * 256;                // float4 chunk id, 4096 total
    int row = idx >> 8;                      // 256 float4 per row
    int c   = (idx & 255) << 2;
    v4f xv = *(const v4f*)(x + ((size_t)(tok0 + row)) * D_DIM + c);
    v4bf o; o[0] = f2bf(xv[0]); o[1] = f2bf(xv[1]); o[2] = f2bf(xv[2]); o[3] = f2bf(xv[3]);
    *(v4bf*)(sX + row * 1032 + c) = o;
  }
  __syncthreads();

  const bf16_t* W = wb + (size_t)mat * D_DIM * D_DIM;
  const int colbase = wid * 128;

  v8f acc[8];
  #pragma unroll
  for (int j = 0; j < 8; ++j) acc[j] = zero8();

  for (int kk = 0; kk < D_DIM; kk += 32) {
    v16bf a = load_a_frag(sX, 1032, kk, lane);
    #pragma unroll
    for (int j = 0; j < 8; ++j) {
      v16bf bfr = load_b_frag(W, D_DIM, colbase + j * 16, kk, lane);
      acc[j] = wmma_bf16(a, bfr, acc[j]);
    }
  }

  const float* bias = (mat == 0) ? bq : (mat == 1) ? bk : bv;
  const float qscale = 0.03125f;             // 1/sqrt(1024), folded into Q

  #pragma unroll
  for (int j = 0; j < 8; ++j) {
    int col = colbase + j * 16 + nl;
    float bval = bias[col];
    if (mat == 2) {
      // V stored transposed: vT[(b*D + col)*S + s]; lane's 8 rows are contiguous s
      int tok  = tok0 + 8 * half;
      int bidx = tok >> 11;                  // / S
      int s    = tok & (S_DIM - 1);
      v8bf o;
      #pragma unroll
      for (int r = 0; r < 8; ++r) o[r] = f2bf(acc[j][r] + bval);
      *(v8bf*)(vT + ((size_t)bidx * D_DIM + col) * S_DIM + s) = o;
    } else {
      bf16_t* dst = (mat == 0) ? qb : kb;
      float scl = (mat == 0) ? qscale : 1.0f;
      #pragma unroll
      for (int r = 0; r < 8; ++r) {
        int tok = tok0 + r + 8 * half;
        dst[(size_t)tok * D_DIM + col] = f2bf((acc[j][r] + bval) * scl);
      }
    }
  }
}

// ---------------------------------------------------------------- kernel 3
// grid: B * S/16. One block = 16 query rows. Each wave owns 16 key tiles and
// keeps ALL score tiles in registers (128 VGPRs): score GEMM runs exactly once,
// softmax is a flat 3-phase reduction over registers (no online rescaling).
__global__ __launch_bounds__(256) void attn_core(
    const bf16_t* __restrict__ qb, const bf16_t* __restrict__ kb,
    const bf16_t* __restrict__ vT, float* __restrict__ out)
{
  const int qt   = blockIdx.x & 127;
  const int b    = blockIdx.x >> 7;
  const int tid  = threadIdx.x;
  const int lane = tid & 31;
  const int w    = tid >> 5;
  const int half = lane >> 4;
  const int nl   = lane & 15;

  __shared__ bf16_t sQ[16 * 1032];           // query tile (bf16)
  __shared__ bf16_t sP[16 * 2056];           // softmax probs, 16 x 2048 (padded)
  __shared__ float  sM[8][16];
  __shared__ float  sS[8][16];
  __shared__ float  sRowMax[16];
  __shared__ float  sRowInv[16];

  const bf16_t* Q = qb + ((size_t)b * S_DIM + qt * 16) * D_DIM;
  const bf16_t* K = kb + (size_t)b * S_DIM * D_DIM;
  const bf16_t* V = vT + (size_t)b * D_DIM * S_DIM;

  // stage Q tile to LDS (already bf16, already pre-scaled by 1/sqrt(D))
  for (int it = 0; it < 8; ++it) {
    int idx = tid + it * 256;                // v8bf chunk id, 2048 total
    int row = idx >> 7;                      // 128 chunks per row
    int c   = (idx & 127) << 3;
    *(v8bf*)(sQ + row * 1032 + c) = *(const v8bf*)(Q + (size_t)row * D_DIM + c);
  }
  __syncthreads();

  // ---- score pass: wave w owns key tiles [w*16, w*16+16).
  // 4 tiles per A-fragment (4 independent WMMA chains); all tiles stay live.
  v8f sc[16];
  #pragma unroll
  for (int g = 0; g < 16; g += 4) {
    const int kt0 = (w * 16 + g) * 16;       // key row of first tile in group
    v8f a0 = zero8(), a1 = zero8(), a2 = zero8(), a3 = zero8();
    for (int kk = 0; kk < D_DIM; kk += 32) {
      v16bf a = load_a_frag(sQ, 1032, kk, lane);
      a0 = wmma_bf16(a, load_b_frag(K, D_DIM, kt0,      kk, lane), a0);
      a1 = wmma_bf16(a, load_b_frag(K, D_DIM, kt0 + 16, kk, lane), a1);
      a2 = wmma_bf16(a, load_b_frag(K, D_DIM, kt0 + 32, kk, lane), a2);
      a3 = wmma_bf16(a, load_b_frag(K, D_DIM, kt0 + 48, kk, lane), a3);
    }
    sc[g] = a0; sc[g + 1] = a1; sc[g + 2] = a2; sc[g + 3] = a3;
  }

  // ---- phase 1: row max (register tree + one 16-lane shuffle reduce per row)
  {
    float rmax[8];
    #pragma unroll
    for (int r = 0; r < 8; ++r) {
      float m = sc[0][r];
      #pragma unroll
      for (int g = 1; g < 16; ++g) m = fmaxf(m, sc[g][r]);
      for (int ms = 1; ms <= 8; ms <<= 1) m = fmaxf(m, __shfl_xor(m, ms, 32));
      rmax[r] = m;
    }
    if (nl == 0) {
      #pragma unroll
      for (int r = 0; r < 8; ++r) sM[w][r + 8 * half] = rmax[r];
    }
  }
  __syncthreads();
  if (tid < 16) {
    float fm = -1e30f;
    for (int ww = 0; ww < 8; ++ww) fm = fmaxf(fm, sM[ww][tid]);
    sRowMax[tid] = fm;
  }
  __syncthreads();

  // ---- phase 2: exp in place + row sum
  {
    float gmax[8];
    #pragma unroll
    for (int r = 0; r < 8; ++r) gmax[r] = sRowMax[r + 8 * half];
    float rsum[8];
    #pragma unroll
    for (int r = 0; r < 8; ++r) {
      float s = 0.f;
      #pragma unroll
      for (int g = 0; g < 16; ++g) {
        float e = __expf(sc[g][r] - gmax[r]);
        sc[g][r] = e;                        // keep exp for the P write
        s += e;
      }
      for (int ms = 1; ms <= 8; ms <<= 1) s += __shfl_xor(s, ms, 32);
      rsum[r] = s;
    }
    if (nl == 0) {
      #pragma unroll
      for (int r = 0; r < 8; ++r) sS[w][r + 8 * half] = rsum[r];
    }
  }
  __syncthreads();
  if (tid < 16) {
    float fs = 0.f;
    for (int ww = 0; ww < 8; ++ww) fs += sS[ww][tid];
    sRowInv[tid] = 1.0f / fs;
  }
  __syncthreads();

  // ---- phase 3: P = exp * inv-sum -> bf16 in LDS
  {
    float ri[8];
    #pragma unroll
    for (int r = 0; r < 8; ++r) ri[r] = sRowInv[r + 8 * half];
    #pragma unroll
    for (int g = 0; g < 16; ++g) {
      const int kcol = (w * 16 + g) * 16 + nl;
      #pragma unroll
      for (int r = 0; r < 8; ++r) {
        sP[(r + 8 * half) * 2056 + kcol] = f2bf(sc[g][r] * ri[r]);
      }
    }
  }
  __syncthreads();

  // ---- out = P @ V  (A-frags from LDS, B-frags from transposed V)
  v8f oacc[8];
  #pragma unroll
  for (int j = 0; j < 8; ++j) oacc[j] = zero8();
  const int colbase = w * 128;

  for (int kk = 0; kk < S_DIM; kk += 32) {
    v16bf a = load_a_frag(sP, 2056, kk, lane);
    #pragma unroll
    for (int j = 0; j < 8; ++j) {
      v16bf bfr = load_b_frag(V, S_DIM, colbase + j * 16, kk, lane);
      oacc[j] = wmma_bf16(a, bfr, oacc[j]);
    }
  }

  float* O = out + ((size_t)b * S_DIM + qt * 16) * D_DIM;
  #pragma unroll
  for (int j = 0; j < 8; ++j) {
    int col = colbase + j * 16 + nl;
    #pragma unroll
    for (int r = 0; r < 8; ++r) {
      O[(size_t)(r + 8 * half) * D_DIM + col] = oacc[j][r];
    }
  }
}

// ---------------------------------------------------------------- launcher
extern "C" void kernel_launch(void* const* d_in, const int* in_sizes, int n_in,
                              void* d_out, int out_size, void* d_ws, size_t ws_size,
                              hipStream_t stream) {
  const float* x  = (const float*)d_in[0];
  const float* Wq = (const float*)d_in[1];
  const float* bq = (const float*)d_in[2];
  const float* Wk = (const float*)d_in[3];
  const float* bk = (const float*)d_in[4];
  const float* Wv = (const float*)d_in[5];
  const float* bv = (const float*)d_in[6];
  float* out = (float*)d_out;

  char* ws = (char*)d_ws;
  const size_t wsz  = (size_t)3 * D_DIM * D_DIM;        // bf16 elements
  const size_t qsz  = (size_t)B_DIM * S_DIM * D_DIM;    // bf16 elements
  bf16_t* wb = (bf16_t*)ws;
  bf16_t* qb = (bf16_t*)(ws + wsz * 2);
  bf16_t* kb = qb + qsz;
  bf16_t* vT = kb + qsz;

  attn_convert_w<<<(D_DIM * D_DIM + 255) / 256, 256, 0, stream>>>(Wq, Wk, Wv, wb);

  dim3 g2(B_DIM * S_DIM / 16, 3);
  attn_qkv<<<g2, 256, 0, stream>>>(x, wb, bq, bk, bv, qb, kb, vT);

  attn_core<<<B_DIM * (S_DIM / 16), 256, 0, stream>>>(qb, kb, vT, out);
}
